// TwoLayerNN_11957188952318
// MI455X (gfx1250) — matrix operand order repaired and checked by reading, not compiled, not run on hardware
//
#include <hip/hip_runtime.h>

typedef __attribute__((ext_vector_type(16))) __bf16 v16bf;
typedef __attribute__((ext_vector_type(8)))  float  v8f;

union Frag { v16bf v; uint4 q[2]; };

__device__ __forceinline__ unsigned short f2bf(float f) {
    unsigned u = __builtin_bit_cast(unsigned, f);
    unsigned r = u + 0x7FFFu + ((u >> 16) & 1u);   // round-to-nearest-even
    return (unsigned short)(r >> 16);
}

// ---------------------------------------------------------------------------
// Prepass 1: W_phi f32 -> bf16 ; pack V = [W_joint(2); W_groups(4); zeros(10)]
// as a 16 x 1024 bf16 matrix (row j = weight vector j).
// ---------------------------------------------------------------------------
__global__ void prep_convert(const float* __restrict__ W_phi,
                             const float* __restrict__ W_joint,
                             const float* __restrict__ W_groups,
                             unsigned short* __restrict__ Wbf,
                             unsigned short* __restrict__ Vbf) {
    int i = blockIdx.x * 256 + threadIdx.x;
    const int NW = 1024 * 1024;
    if (i < NW) {
        Wbf[i] = f2bf(W_phi[i]);
    } else {
        int k = i - NW;
        if (k < 16 * 1024) {
            int j = k >> 10, h = k & 1023;
            float val = 0.0f;
            if (j < 2)      val = W_joint[j * 1024 + h];
            else if (j < 6) val = W_groups[(j - 2) * 1024 + h];
            Vbf[k] = f2bf(val);
        }
    }
}

// ---------------------------------------------------------------------------
// Prepass 2: bphiV[j] = dot(b_phi, V[j]) for j in 0..5
// ---------------------------------------------------------------------------
__global__ void prep_bphiv(const float* __restrict__ b_phi,
                           const float* __restrict__ W_joint,
                           const float* __restrict__ W_groups,
                           float* __restrict__ bphiV) {
    int wave = threadIdx.x >> 5, lane = threadIdx.x & 31;
    if (wave >= 6) return;
    const float* v = (wave < 2) ? (W_joint + wave * 1024)
                                : (W_groups + (wave - 2) * 1024);
    float s = 0.0f;
    for (int k = lane; k < 1024; k += 32) s += b_phi[k] * v[k];
    for (int m = 16; m >= 1; m >>= 1) s += __shfl_xor(s, m, 32);
    if (lane == 0) bphiV[wave] = s;
}

// ---------------------------------------------------------------------------
// Fused: phi = X @ W_phi^T (+b_phi folded), out6 = phi @ V^T, gather, store.
// Grid: B/128 workgroups of 256 threads (8 waves). Wave w owns phi cols
// [128w, 128w+128). Stage-1 register blocking: 4M x 2N per wave.
// Stage-2 runs on WMMA via a per-wave LDS bf16 scratch (A-layout reload).
// ---------------------------------------------------------------------------
#define XS_STRIDE 1032          // 1024 + 8 bf16 pad -> conflict-free A frags
#define SC_STRIDE 40            // 32 + 8 bf16 pad   -> conflict-free A2 frags

__global__ __launch_bounds__(256) void fused_kernel(
    const float* __restrict__ X,
    const unsigned short* __restrict__ Wbf,
    const unsigned short* __restrict__ Vbf,
    const float* __restrict__ bphiV,
    const float* __restrict__ b_groups,
    const float* __restrict__ b_joint,
    const int* __restrict__ D,
    const int* __restrict__ Dagn,
    float* __restrict__ out, int nB)
{
    __shared__ __align__(16) unsigned short Xs[128 * XS_STRIDE];    // 264192 B
    __shared__ __align__(16) unsigned short Sc[8 * 64 * SC_STRIDE]; //  40960 B

    const int t    = threadIdx.x;
    const int row0 = blockIdx.x * 128;

    // ---- cooperative X tile load + bf16 convert into LDS ----
    #pragma unroll 4
    for (int i = 0; i < 128; ++i) {
        int vid = i * 256 + t;          // vec4 id within 128x1024 tile
        int r = vid >> 8, kv = vid & 255;
        float4 f = ((const float4*)(X + (size_t)(row0 + r) * 1024))[kv];
        unsigned long long p =
            (unsigned long long)f2bf(f.x)        |
            ((unsigned long long)f2bf(f.y) << 16)|
            ((unsigned long long)f2bf(f.z) << 32)|
            ((unsigned long long)f2bf(f.w) << 48);
        *(unsigned long long*)&Xs[r * XS_STRIDE + kv * 4] = p;
    }
    __syncthreads();

    const int wave = t >> 5, lane = t & 31;
    const int hs = lane >> 4, l = lane & 15;
    const int colBase = wave * 128;
    unsigned short* sw = &Sc[wave * 64 * SC_STRIDE];

    v8f c2[8];                       // out6 accum: 8 M-tiles x (16 rows x 16 j)
    #pragma unroll
    for (int mt = 0; mt < 8; ++mt) c2[mt] = (v8f)0.0f;

    for (int ntp = 0; ntp < 4; ++ntp) {        // 32-col chunk of this wave
        // stage-2 B frag: V^T chunk for this 32-col slice (k2 = 32)
        Frag B2;
        B2.q[0] = *(const uint4*)(Vbf + (size_t)l * 1024 + colBase + ntp * 32 + hs * 8);
        B2.q[1] = *(const uint4*)(Vbf + (size_t)l * 1024 + colBase + ntp * 32 + 16 + hs * 8);

        #pragma unroll
        for (int mtg = 0; mtg < 2; ++mtg) {    // 64-row half of the tile
            v8f acc[4][2];
            #pragma unroll
            for (int mt = 0; mt < 4; ++mt)
                #pragma unroll
                for (int nt = 0; nt < 2; ++nt) acc[mt][nt] = (v8f)0.0f;

            #pragma unroll 4
            for (int k = 0; k < 32; ++k) {     // K = 1024 in steps of 32
                Frag A[4];
                #pragma unroll
                for (int mt = 0; mt < 4; ++mt) {
                    int row = mtg * 64 + mt * 16 + l;
                    const uint4* p =
                        (const uint4*)&Xs[row * XS_STRIDE + k * 32 + hs * 8];
                    A[mt].q[0] = p[0];
                    A[mt].q[1] = p[2];         // +16 elems (second K-run)
                }
                Frag Bf[2];
                #pragma unroll
                for (int nt = 0; nt < 2; ++nt) {
                    int n = colBase + ntp * 32 + nt * 16 + l;
                    const uint4* pb = (const uint4*)(Wbf + (size_t)n * 1024 + k * 32 + hs * 8);
                    Bf[nt].q[0] = pb[0];
                    Bf[nt].q[1] = pb[2];
                }
                #pragma unroll
                for (int mt = 0; mt < 4; ++mt)
                    #pragma unroll
                    for (int nt = 0; nt < 2; ++nt)
                        acc[mt][nt] = __builtin_amdgcn_wmma_f32_16x16x32_bf16(
                            false, A[mt].v, false, Bf[nt].v,
                            (short)0, acc[mt][nt], false, false);
            }

            // phi block (64 rows x 32 cols) -> bf16 scratch, row-major
            #pragma unroll
            for (int mt = 0; mt < 4; ++mt)
                #pragma unroll
                for (int nt = 0; nt < 2; ++nt)
                    #pragma unroll
                    for (int r = 0; r < 8; ++r) {
                        int rl  = mt * 16 + hs * 8 + r;
                        int col = nt * 16 + l;
                        sw[rl * SC_STRIDE + col] = f2bf(acc[mt][nt][r]);
                    }

            // stage-2 WMMA: out6 += phi_block @ V^T_chunk
            #pragma unroll
            for (int mtl = 0; mtl < 4; ++mtl) {
                Frag A2;
                int rl = mtl * 16 + l;
                const uint4* p = (const uint4*)&sw[rl * SC_STRIDE + hs * 8];
                A2.q[0] = p[0];
                A2.q[1] = p[2];
                c2[mtg * 4 + mtl] = __builtin_amdgcn_wmma_f32_16x16x32_bf16(
                    false, A2.v, false, B2.v,
                    (short)0, c2[mtg * 4 + mtl], false, false);
            }
        }
    }

    // ---- cross-wave reduction of out6 partials (reuse Xs as f32 buffer) ----
    __syncthreads();
    float* red = (float*)Xs;          // [wave][mt][r][lane] = 8*8*8*32 f32
    #pragma unroll
    for (int mt = 0; mt < 8; ++mt)
        #pragma unroll
        for (int r = 0; r < 8; ++r)
            red[((wave * 8 + mt) * 8 + r) * 32 + lane] = c2[mt][r];
    __syncthreads();

    // wave w finalizes M-tile mt = w
    const int j = l;                  // output column 0..5 valid
    float bv = 0.0f, bj = 0.0f; int g = 0, cc = 0;
    if (j < 6) bv = bphiV[j];
    if (j < 2) bj = b_joint[j];
    else if (j < 6) { g = (j - 2) >> 1; cc = (j - 2) & 1; bj = b_groups[j - 2]; }
    const size_t seg = (size_t)nB * 2;

    #pragma unroll
    for (int r = 0; r < 8; ++r) {
        float s = 0.0f;
        #pragma unroll
        for (int w2 = 0; w2 < 8; ++w2)
            s += red[((w2 * 8 + wave) * 8 + r) * 32 + lane];
        int row = row0 + wave * 16 + hs * 8 + r;
        float v = s + bv + bj;
        if (j < 2) {
            out[(size_t)row * 2 + j] = v;                       // joint
        } else if (j < 6) {
            if (D[row]    == g) out[seg     + (size_t)row * 2 + cc] = v; // specific
            if (Dagn[row] == g) out[2 * seg + (size_t)row * 2 + cc] = v; // agnostic
        }
    }
}

// ---------------------------------------------------------------------------
extern "C" void kernel_launch(void* const* d_in, const int* in_sizes, int n_in,
                              void* d_out, int out_size, void* d_ws, size_t ws_size,
                              hipStream_t stream) {
    const float* X        = (const float*)d_in[0];
    const float* W_phi    = (const float*)d_in[1];
    const float* b_phi    = (const float*)d_in[2];
    const float* W_groups = (const float*)d_in[3];
    const float* b_groups = (const float*)d_in[4];
    const float* W_joint  = (const float*)d_in[5];
    const float* b_joint  = (const float*)d_in[6];
    const int*   D        = (const int*)d_in[7];
    const int*   Dagn     = (const int*)d_in[8];
    float* out = (float*)d_out;
    const int nB = in_sizes[7];               // 65536

    // ws layout: Wbf (1M bf16) | Vbf (16K bf16) | bphiV (6 f32)  ~= 2.1 MB
    unsigned short* Wbf = (unsigned short*)d_ws;
    unsigned short* Vbf = Wbf + 1024 * 1024;
    float* bphiV = (float*)(Vbf + 16 * 1024);

    int Nconv = 1024 * 1024 + 16 * 1024;
    prep_convert<<<(Nconv + 255) / 256, 256, 0, stream>>>(W_phi, W_joint, W_groups, Wbf, Vbf);
    prep_bphiv<<<1, 256, 0, stream>>>(b_phi, W_joint, W_groups, bphiV);
    fused_kernel<<<nB / 128, 256, 0, stream>>>(X, Wbf, Vbf, bphiV,
                                               b_groups, b_joint, D, Dagn, out, nB);
}
